// DiTBlock_4758823763996
// MI455X (gfx1250) — matrix operand order repaired and checked by reading, not compiled, not run on hardware
//
#include <hip/hip_runtime.h>
#include <hip/hip_bf16.h>

typedef _Float16 v16h __attribute__((ext_vector_type(16)));
typedef _Float16 v8h  __attribute__((ext_vector_type(8)));
typedef float    v8f  __attribute__((ext_vector_type(8)));

union HV { v16h v; v8h h[2]; };

// ---- CDNA5 async global->LDS (ASYNCcnt-tracked DMA into LDS, no VGPR data) ----
__device__ __forceinline__ void async_ld16(unsigned lds_off, const _Float16* gaddr) {
    asm volatile("global_load_async_to_lds_b128 %0, %1, off"
                 :: "v"(lds_off), "v"(gaddr) : "memory");
}
__device__ __forceinline__ void wait_async0() {
    asm volatile("s_wait_asynccnt 0" ::: "memory");
}
__device__ __forceinline__ unsigned lds_addr(const void* p) {
    return (unsigned)(uintptr_t)p;   // low 32 bits of flat LDS pointer = wave LDS offset
}

// ---------------------------------------------------------------- cvt f32->f16
__global__ __launch_bounds__(256)
void cvt_kernel(const float* __restrict__ in, _Float16* __restrict__ out, int n4) {
    int i = blockIdx.x * 256 + threadIdx.x;
    if (i < n4) {
        float4 v = ((const float4*)in)[i];
        _Float16* o = out + (size_t)i * 4;
        o[0] = (_Float16)v.x; o[1] = (_Float16)v.y;
        o[2] = (_Float16)v.z; o[3] = (_Float16)v.w;
    }
}

// ---------------------------------------------------------------- LayerNorm (C=1024, 256 thr)
__global__ __launch_bounds__(256)
void ln_kernel(const float* __restrict__ x, const float* __restrict__ g,
               const float* __restrict__ bta, _Float16* __restrict__ out, int C) {
    int row = blockIdx.x;
    int lane = threadIdx.x & 31, wid = threadIdx.x >> 5;
    const float4 v = ((const float4*)(x + (size_t)row * C))[threadIdx.x];
    __shared__ float red[8];
    float s = v.x + v.y + v.z + v.w;
#pragma unroll
    for (int off = 16; off; off >>= 1) s += __shfl_xor(s, off, 32);
    if (lane == 0) red[wid] = s;
    __syncthreads();
    float tot = 0.f;
#pragma unroll
    for (int i = 0; i < 8; ++i) tot += red[i];
    float mean = tot / (float)C;
    float dx = v.x - mean, dy = v.y - mean, dz = v.z - mean, dw = v.w - mean;
    float s2 = dx * dx + dy * dy + dz * dz + dw * dw;
#pragma unroll
    for (int off = 16; off; off >>= 1) s2 += __shfl_xor(s2, off, 32);
    __syncthreads();
    if (lane == 0) red[wid] = s2;
    __syncthreads();
    float tv = 0.f;
#pragma unroll
    for (int i = 0; i < 8; ++i) tv += red[i];
    float rstd = rsqrtf(tv / (float)C + 1e-6f);
    int c0 = threadIdx.x * 4;
    const float4 gg = *(const float4*)(g + c0);
    const float4 bb = *(const float4*)(bta + c0);
    _Float16* o = out + (size_t)row * C + c0;
    o[0] = (_Float16)(dx * rstd * gg.x + bb.x);
    o[1] = (_Float16)(dy * rstd * gg.y + bb.y);
    o[2] = (_Float16)(dz * rstd * gg.z + bb.z);
    o[3] = (_Float16)(dw * rstd * gg.w + bb.w);
}

// ---------------------------------------------------------------- WMMA GEMM
// out = act(A @ W^T + bias) (+resid). A:(M,K) f16 rm, W:(N,K) f16 rm.
// Block tile 128x128, BK=64, 128 thr = 4 waves, each wave a 64x64 patch
// (4x4 WMMA tiles, 128 accum VGPRs). Async double-buffered LDS staging:
// 32 WMMAs per 32 ds_load_b128 per stage, one barrier per stage.
template <bool OUT16, bool DOGELU, bool RESID>
__global__ __launch_bounds__(128)
void gemm_kernel(const _Float16* __restrict__ A, const _Float16* __restrict__ W,
                 const float* __restrict__ bias, const float* __restrict__ resid,
                 void* __restrict__ outv, int M, int N, int K) {
    __shared__ alignas(16) _Float16 As[2][128][72];
    __shared__ alignas(16) _Float16 Bs[2][128][72];
    const int m0 = blockIdx.y * 128, n0 = blockIdx.x * 128;
    const int t = threadIdx.x;
    const int lane = t & 31, wid = t >> 5;
    const int wm = (wid >> 1) * 64, wn = (wid & 1) * 64;
    const int mrow = lane & 15, khalf = lane >> 4;

    // staging: thread t owns row t of both tiles (64 halves = 8 async B128s each)
    auto stage = [&](int kk, int buf) {
        const _Float16* ga = A + (size_t)(m0 + t) * K + kk;
        const _Float16* gb = W + (size_t)(n0 + t) * K + kk;
        unsigned la = lds_addr(&As[buf][t][0]);
        unsigned lb = lds_addr(&Bs[buf][t][0]);
#pragma unroll
        for (int j = 0; j < 8; ++j) {
            async_ld16(la + j * 16, ga + j * 8);
            async_ld16(lb + j * 16, gb + j * 8);
        }
    };

    v8f acc[4][4] = {};
    const int nk = K >> 6;
    stage(0, 0);
    for (int kb = 0; kb < nk; ++kb) {
        wait_async0();
        __syncthreads();
        if (kb + 1 < nk) stage((kb + 1) << 6, (kb + 1) & 1);
        const int buf = kb & 1;
#pragma unroll
        for (int ks = 0; ks < 2; ++ks) {
            v16h a[4];
#pragma unroll
            for (int mt = 0; mt < 4; ++mt) {
                HV u;
                u.h[0] = *(const v8h*)&As[buf][wm + mt * 16 + mrow][ks * 32 + khalf * 8];
                u.h[1] = *(const v8h*)&As[buf][wm + mt * 16 + mrow][ks * 32 + khalf * 8 + 16];
                a[mt] = u.v;
            }
#pragma unroll
            for (int nt = 0; nt < 4; ++nt) {
                HV u;
                u.h[0] = *(const v8h*)&Bs[buf][wn + nt * 16 + mrow][ks * 32 + khalf * 8];
                u.h[1] = *(const v8h*)&Bs[buf][wn + nt * 16 + mrow][ks * 32 + khalf * 8 + 16];
#pragma unroll
                for (int mt = 0; mt < 4; ++mt)
                    acc[mt][nt] = __builtin_amdgcn_wmma_f32_16x16x32_f16(
                        false, a[mt], false, u.v, (short)0, acc[mt][nt], false, false);
            }
        }
    }

#pragma unroll
    for (int nt = 0; nt < 4; ++nt) {
        int col = n0 + wn + nt * 16 + mrow;
        float bz = bias[col];
#pragma unroll
        for (int mt = 0; mt < 4; ++mt) {
#pragma unroll
            for (int r = 0; r < 8; ++r) {
                int row = m0 + wm + mt * 16 + r + khalf * 8;
                float v = acc[mt][nt][r] + bz;
                if constexpr (DOGELU) v = 0.5f * v * (1.0f + erff(v * 0.70710678118f));
                if constexpr (RESID)  v += resid[(size_t)row * N + col];
                if constexpr (OUT16)  ((_Float16*)outv)[(size_t)row * N + col] = (_Float16)v;
                else                  ((float*)outv)[(size_t)row * N + col] = v;
            }
        }
    }
}

// ---------------------------------------------------------------- RoPE (q,k) from fused qkv
// qkv: (B,N,3C) f16.  qr/kr: (B,H,N,D) f16.
__global__ __launch_bounds__(256)
void rope_kernel(const _Float16* __restrict__ qkv, const float* __restrict__ cs,
                 const float* __restrict__ sn, _Float16* __restrict__ qr,
                 _Float16* __restrict__ kr, int N, int H, int C) {
    const int D = 64;
    int tid = blockIdx.x * 256 + threadIdx.x;       // B*H*N*32 threads
    int d  = tid & 31;
    int n  = (tid >> 5) % N;
    int h  = ((tid >> 5) / N) % H;
    int b  = (tid >> 5) / (N * H);
    int C3 = 3 * C;
    size_t qsrc = ((size_t)b * N + n) * C3 + h * D + d;
    size_t ksrc = qsrc + C;
    size_t dst  = (((size_t)b * H + h) * N + n) * D + d;
    float c1 = cs[n * D + d],      s1 = sn[n * D + d];
    float c2 = cs[n * D + d + 32], s2 = sn[n * D + d + 32];
    float q1 = (float)qkv[qsrc], q2 = (float)qkv[qsrc + 32];
    qr[dst]      = (_Float16)(q1 * c1 - q2 * s1);
    qr[dst + 32] = (_Float16)(q2 * c2 + q1 * s2);
    float k1 = (float)qkv[ksrc], k2 = (float)qkv[ksrc + 32];
    kr[dst]      = (_Float16)(k1 * c1 - k2 * s1);
    kr[dst + 32] = (_Float16)(k2 * c2 + k1 * s2);
}

// ---------------------------------------------------------------- V transpose via LDS tile
// qkv v-slice (B,N,3C)[.., 2C + h*64 + d]  ->  vt: (B*H, 64, N) f16
__global__ __launch_bounds__(256)
void vtrans_kernel(const _Float16* __restrict__ qkv, _Float16* __restrict__ vt,
                   int N, int H, int C) {
    __shared__ alignas(16) _Float16 T[64][72];
    const int D = 64, C3 = 3 * C;
    int bh = blockIdx.x, b = bh / H, h = bh % H;
    int n0 = blockIdx.y * 64;
    int t = threadIdx.x;
    {
        int nr = t >> 2, c0 = (t & 3) * 16;
        const uint4* src = (const uint4*)(qkv + ((size_t)b * N + n0 + nr) * C3 + 2 * C + h * D + c0);
        *(uint4*)&T[nr][c0]     = src[0];
        *(uint4*)&T[nr][c0 + 8] = src[1];
    }
    __syncthreads();
    int d = t >> 2, nb = (t & 3) * 16;
    _Float16 tmp[16];
#pragma unroll
    for (int i = 0; i < 16; ++i) tmp[i] = T[nb + i][d];
    uint4* dst = (uint4*)(vt + ((size_t)bh * D + d) * N + n0 + nb);
    dst[0] = ((uint4*)tmp)[0];
    dst[1] = ((uint4*)tmp)[1];
}

// ---------------------------------------------------------------- flash attention
// Q,Kh: (B*H, N, 64) f16; Vt: (B*H, 64, N) f16; O: (B, N, H*64) f16.
// grid (N/64, B*H), 128 thr = 4 waves; async double-buffered K/V staging.
__global__ __launch_bounds__(128)
void flash_attn_kernel(const _Float16* __restrict__ Q, const _Float16* __restrict__ Kh,
                       const _Float16* __restrict__ Vt, _Float16* __restrict__ O,
                       int N, int H) {
    const int D = 64;
    int bh = blockIdx.y;
    int b = bh / H, h = bh % H;
    int t = threadIdx.x;
    int lane = t & 31, wid = t >> 5;
    int mrow = lane & 15, khalf = lane >> 4;
    int q0 = blockIdx.x * 64;

    __shared__ alignas(16) _Float16 Ks[2][64][72];
    __shared__ alignas(16) _Float16 Vs[2][64][72];
    __shared__ alignas(16) _Float16 Ps[4][16][72];

    // Q fragments live in registers for the whole kernel
    const _Float16* qptr = Q + ((size_t)bh * N + q0 + wid * 16 + mrow) * D;
    HV u0, u1;
    u0.h[0] = *(const v8h*)(qptr + khalf * 8);
    u0.h[1] = *(const v8h*)(qptr + khalf * 8 + 16);
    u1.h[0] = *(const v8h*)(qptr + 32 + khalf * 8);
    u1.h[1] = *(const v8h*)(qptr + 32 + khalf * 8 + 16);
    v16h aq0 = u0.v, aq1 = u1.v;

    const int srow = t >> 1, scol = (t & 1) * 32;
    auto stage = [&](int kc, int buf) {
        const _Float16* gk = Kh + ((size_t)bh * N + kc + srow) * D + scol;
        const _Float16* gv = Vt + ((size_t)bh * D + srow) * N + kc + scol;
        unsigned lk = lds_addr(&Ks[buf][srow][scol]);
        unsigned lv = lds_addr(&Vs[buf][srow][scol]);
#pragma unroll
        for (int j = 0; j < 4; ++j) {
            async_ld16(lk + j * 16, gk + j * 8);
            async_ld16(lv + j * 16, gv + j * 8);
        }
    };

    float m[8], l[8];
    v8f oa[4] = {};
#pragma unroll
    for (int r = 0; r < 8; ++r) { m[r] = -1e30f; l[r] = 0.f; }

    stage(0, 0);
    for (int kc = 0, it = 0; kc < N; kc += 64, ++it) {
        wait_async0();
        __syncthreads();
        if (kc + 64 < N) stage(kc + 64, (it + 1) & 1);
        const int buf = it & 1;

        // S = Q K^T for this wave's 16 rows x 64 keys
        v8f s[4] = {};
#pragma unroll
        for (int nt = 0; nt < 4; ++nt) {
            HV b0, b1;
            int krow = nt * 16 + mrow;
            b0.h[0] = *(const v8h*)&Ks[buf][krow][khalf * 8];
            b0.h[1] = *(const v8h*)&Ks[buf][krow][khalf * 8 + 16];
            b1.h[0] = *(const v8h*)&Ks[buf][krow][32 + khalf * 8];
            b1.h[1] = *(const v8h*)&Ks[buf][krow][32 + khalf * 8 + 16];
            s[nt] = __builtin_amdgcn_wmma_f32_16x16x32_f16(false, aq0, false, b0.v, (short)0, s[nt], false, false);
            s[nt] = __builtin_amdgcn_wmma_f32_16x16x32_f16(false, aq1, false, b1.v, (short)0, s[nt], false, false);
        }

        // online softmax (row r + 8*khalf lives within a 16-lane half)
#pragma unroll
        for (int r = 0; r < 8; ++r) {
            float mx = -1e30f;
#pragma unroll
            for (int nt = 0; nt < 4; ++nt) { s[nt][r] *= 0.125f; mx = fmaxf(mx, s[nt][r]); }
#pragma unroll
            for (int off = 1; off < 16; off <<= 1) mx = fmaxf(mx, __shfl_xor(mx, off, 32));
            float mnew = fmaxf(m[r], mx);
            float alpha = __expf(m[r] - mnew);
            m[r] = mnew;
            float rs = 0.f;
#pragma unroll
            for (int nt = 0; nt < 4; ++nt) {
                float p = __expf(s[nt][r] - mnew);
                s[nt][r] = p;
                rs += p;
                oa[nt][r] *= alpha;
            }
#pragma unroll
            for (int off = 1; off < 16; off <<= 1) rs += __shfl_xor(rs, off, 32);
            l[r] = l[r] * alpha + rs;
        }

        // P tile -> wave-local LDS (C/D layout -> A fragment layout)
#pragma unroll
        for (int nt = 0; nt < 4; ++nt)
#pragma unroll
            for (int r = 0; r < 8; ++r)
                Ps[wid][r + khalf * 8][nt * 16 + mrow] = (_Float16)s[nt][r];

        // O += P @ V   (K = 64 keys, two k-steps)
#pragma unroll
        for (int ks = 0; ks < 2; ++ks) {
            HV ua;
            ua.h[0] = *(const v8h*)&Ps[wid][mrow][ks * 32 + khalf * 8];
            ua.h[1] = *(const v8h*)&Ps[wid][mrow][ks * 32 + khalf * 8 + 16];
#pragma unroll
            for (int nt = 0; nt < 4; ++nt) {
                HV ub;
                int drow = nt * 16 + mrow;
                ub.h[0] = *(const v8h*)&Vs[buf][drow][ks * 32 + khalf * 8];
                ub.h[1] = *(const v8h*)&Vs[buf][drow][ks * 32 + khalf * 8 + 16];
                oa[nt] = __builtin_amdgcn_wmma_f32_16x16x32_f16(false, ua.v, false, ub.v, (short)0, oa[nt], false, false);
            }
        }
    }

    const int C = H * D;
#pragma unroll
    for (int nt = 0; nt < 4; ++nt) {
        int d = nt * 16 + mrow;
#pragma unroll
        for (int r = 0; r < 8; ++r) {
            int row = q0 + wid * 16 + r + khalf * 8;
            float v = oa[nt][r] / l[r];
            O[((size_t)b * N + row) * C + h * D + d] = (_Float16)v;
        }
    }
}

// ---------------------------------------------------------------- launch
extern "C" void kernel_launch(void* const* d_in, const int* in_sizes, int n_in,
                              void* d_out, int out_size, void* d_ws, size_t ws_size,
                              hipStream_t stream) {
    constexpr int B = 2, N = 2048, C = 1024, H = 16, F = 4096;
    constexpr int M = B * N;
    (void)in_sizes; (void)n_in; (void)out_size; (void)ws_size;

    const float* x   = (const float*)d_in[0];
    const float* rc  = (const float*)d_in[1];
    const float* rs  = (const float*)d_in[2];
    const float* g1  = (const float*)d_in[3];
    const float* be1 = (const float*)d_in[4];
    const float* Wq  = (const float*)d_in[5];
    const float* bq  = (const float*)d_in[6];
    const float* Wk  = (const float*)d_in[7];
    const float* bk  = (const float*)d_in[8];
    const float* Wv  = (const float*)d_in[9];
    const float* bv  = (const float*)d_in[10];
    const float* Wo  = (const float*)d_in[11];
    const float* bo  = (const float*)d_in[12];
    const float* g2  = (const float*)d_in[13];
    const float* be2 = (const float*)d_in[14];
    const float* W1  = (const float*)d_in[15];
    const float* b1  = (const float*)d_in[16];
    const float* W2  = (const float*)d_in[17];
    const float* b2  = (const float*)d_in[18];

    char* w = (char*)d_ws;
    auto take = [&](size_t bytes) -> char* {
        char* p = w; w += (bytes + 255) & ~(size_t)255; return p;
    };
    _Float16* h1     = (_Float16*)take((size_t)M * C * 2);
    _Float16* qkvf   = (_Float16*)take((size_t)M * 3 * C * 2);
    _Float16* qr     = (_Float16*)take((size_t)M * C * 2);
    _Float16* kr     = (_Float16*)take((size_t)M * C * 2);
    _Float16* vt     = (_Float16*)take((size_t)M * C * 2);
    _Float16* aout   = (_Float16*)take((size_t)M * C * 2);
    float*    x1     = (float*)   take((size_t)M * C * 4);
    _Float16* h2     = (_Float16*)take((size_t)M * C * 2);
    _Float16* h3     = (_Float16*)take((size_t)M * F * 2);
    _Float16* Wqkv16 = (_Float16*)take((size_t)3 * C * C * 2);
    _Float16* Wo16   = (_Float16*)take((size_t)C * C * 2);
    _Float16* W116   = (_Float16*)take((size_t)F * C * 2);
    _Float16* W216   = (_Float16*)take((size_t)C * F * 2);
    float*    bqkv   = (float*)   take((size_t)3 * C * 4);

    // 1. weight conversion fp32 -> f16 (Wq|Wk|Wv fused contiguously)
    cvt_kernel<<<C * C / 4 / 256, 256, 0, stream>>>(Wq, Wqkv16, C * C / 4);
    cvt_kernel<<<C * C / 4 / 256, 256, 0, stream>>>(Wk, Wqkv16 + (size_t)C * C, C * C / 4);
    cvt_kernel<<<C * C / 4 / 256, 256, 0, stream>>>(Wv, Wqkv16 + (size_t)2 * C * C, C * C / 4);
    cvt_kernel<<<C * C / 4 / 256, 256, 0, stream>>>(Wo, Wo16, C * C / 4);
    cvt_kernel<<<F * C / 4 / 256, 256, 0, stream>>>(W1, W116, F * C / 4);
    cvt_kernel<<<C * F / 4 / 256, 256, 0, stream>>>(W2, W216, C * F / 4);
    hipMemcpyAsync(bqkv,         bq, C * sizeof(float), hipMemcpyDeviceToDevice, stream);
    hipMemcpyAsync(bqkv + C,     bk, C * sizeof(float), hipMemcpyDeviceToDevice, stream);
    hipMemcpyAsync(bqkv + 2 * C, bv, C * sizeof(float), hipMemcpyDeviceToDevice, stream);

    // 2. LN1
    ln_kernel<<<M, 256, 0, stream>>>(x, g1, be1, h1, C);

    // 3. fused QKV projection: (M,C) @ (3C,C)^T -> (M,3C)
    gemm_kernel<true, false, false><<<dim3(3 * C / 128, M / 128), 128, 0, stream>>>(
        h1, Wqkv16, bqkv, nullptr, qkvf, M, 3 * C, C);

    // 4. RoPE (q,k) + V transpose
    rope_kernel<<<(B * H * N * 32) / 256, 256, 0, stream>>>(qkvf, rc, rs, qr, kr, N, H, C);
    vtrans_kernel<<<dim3(B * H, N / 64), 256, 0, stream>>>(qkvf, vt, N, H, C);

    // 5. flash attention
    flash_attn_kernel<<<dim3(N / 64, B * H), 128, 0, stream>>>(qr, kr, vt, aout, N, H);

    // 6. output projection + residual -> x1 (fp32)
    gemm_kernel<false, false, true><<<dim3(C / 128, M / 128), 128, 0, stream>>>(
        aout, Wo16, bo, x, (void*)x1, M, C, C);

    // 7. LN2
    ln_kernel<<<M, 256, 0, stream>>>(x1, g2, be2, h2, C);

    // 8. FFN1 + exact GELU
    gemm_kernel<true, true, false><<<dim3(F / 128, M / 128), 128, 0, stream>>>(
        h2, W116, b1, nullptr, h3, M, F, C);

    // 9. FFN2 + residual -> d_out (fp32)
    gemm_kernel<false, false, true><<<dim3(C / 128, M / 128), 128, 0, stream>>>(
        h3, W216, b2, x1, d_out, M, C, F);
}
